// DeepIRTModel_54030688584005
// MI455X (gfx1250) — matrix operand order repaired and checked by reading, not compiled, not run on hardware
//
#include <hip/hip_runtime.h>
#include <math.h>

#define Bsz 2048
#define Ssz 200
#define Msz 50
#define Ksz 50
#define Vsz 200
#define Fsz 50
#define TB  16          // batch tile per workgroup (== WMMA M)
#define NT  512         // threads per block (16 wave32)
#define KPAD 224        // V (=GEMM K dim) padded to 7 x 32
#define NPAD 208        // V (=GEMM N dim) padded to 13 x 16

typedef __attribute__((ext_vector_type(4)))  float   f4;
typedef __attribute__((ext_vector_type(8)))  float   v8f;
typedef __attribute__((ext_vector_type(4)))  __bf16  v4bf;
typedef __attribute__((ext_vector_type(8)))  __bf16  v8bf;
typedef __attribute__((ext_vector_type(16))) __bf16  v16bf;

// dynamic-LDS layout (bytes)
#define OFF_WT    0                                   // __bf16 [2*NPAD][KPAD] transposed gates
#define SZ_WT     (2 * NPAD * KPAD * 2)               // 186368
#define OFF_QAB   (OFF_WT + SZ_WT)                    // __bf16 [TB][KPAD]
#define SZ_QAB    (TB * KPAD * 2)                     // 7168
#define OFF_SUMW  (OFF_QAB + SZ_QAB)                  // float [250*50]
#define SZ_SUMW   ((Vsz + Ksz) * Fsz * 4)             // 50000
#define OFF_KM    (OFF_SUMW + SZ_SUMW)                // float [50*50]
#define SZ_KM     (Msz * Ksz * 4)                     // 10000
#define DYN_SMEM  (OFF_KM + SZ_KM)                    // 253536

__device__ __forceinline__ float sigmoidf_(float x) { return 1.0f / (1.0f + __expf(-x)); }
__device__ __forceinline__ float softplusf_(float x) { return (x > 20.0f) ? x : log1pf(__expf(x)); }

__global__ __launch_bounds__(NT) void deep_irt_kernel(
    const int*   __restrict__ q_data,  const int*   __restrict__ qa_data,
    const float* __restrict__ q_embed_w, const float* __restrict__ qa_embed_w,
    const float* __restrict__ key_memory, const float* __restrict__ init_value_memory,
    const float* __restrict__ erase_W, const float* __restrict__ erase_b,
    const float* __restrict__ add_W,   const float* __restrict__ add_b,
    const float* __restrict__ summary_W, const float* __restrict__ summary_b,
    const float* __restrict__ theta_W, const float* __restrict__ theta_b,
    const float* __restrict__ beta_W,  const float* __restrict__ beta_b,
    const float* __restrict__ alpha_W, const float* __restrict__ alpha_b,
    float* __restrict__ Mv,            // [B, M*V] scratch (82 MB, L2-resident)
    float* __restrict__ out)           // p|theta|beta|alpha|z, each [B,S]
{
    extern __shared__ char smem_dyn[];
    __bf16* s_Wt   = (__bf16*)(smem_dyn + OFF_WT);    // [(g*NPAD+col)*KPAD + k], zero-padded
    __bf16* s_qab  = (__bf16*)(smem_dyn + OFF_QAB);   // [b*KPAD + k], zero-padded
    float*  s_sumW = (float*) (smem_dyn + OFF_SUMW);
    float*  s_km   = (float*) (smem_dyn + OFF_KM);

    __shared__ float s_qe[TB][Ksz];
    __shared__ float s_w [TB][Msz];
    __shared__ float s_e [TB][Vsz];
    __shared__ float s_a [TB][Vsz];
    __shared__ float s_rd[TB][Vsz];
    __shared__ float s_s [TB][Fsz];
    __shared__ int   s_iq[TB], s_iqa[TB];

    const int tid     = threadIdx.x;
    const int lane    = tid & 31;
    const int wave    = tid >> 5;
    const int nwaves  = NT >> 5;
    const int b0      = blockIdx.x * TB;
    const int halfSel = lane >> 4;      // 0: lanes 0-15, 1: lanes 16-31
    const int l16     = lane & 15;

    // ---- one-time staging: gate weights (bf16, transposed, zero-padded) ----
    for (int p = tid; p < 2 * NPAD * KPAD; p += NT) {
        int g = p / (NPAD * KPAD), r = p % (NPAD * KPAD);
        int col = r / KPAD, k = r % KPAD;
        const float* W = g ? add_W : erase_W;
        float val = (col < Vsz && k < Vsz) ? W[k * Vsz + col] : 0.f;
        s_Wt[p] = (__bf16)val;
    }
    for (int p = tid; p < (Vsz + Ksz) * Fsz; p += NT) s_sumW[p] = summary_W[p];
    for (int p = tid; p < Msz * Ksz; p += NT)         s_km[p]  = key_memory[p];
    for (int p = tid; p < TB * KPAD; p += NT)         s_qab[p] = (__bf16)0.f;
    // ---- init this tile's Mv slice (fresh every launch: deterministic) ----
    for (int p = tid; p < TB * Msz * Vsz; p += NT) {
        int b = p / (Msz * Vsz), r = p % (Msz * Vsz);
        Mv[(size_t)(b0 + b) * (Msz * Vsz) + r] = init_value_memory[r];
    }
    __syncthreads();

    for (int t = 0; t < Ssz; ++t) {
        if (tid < TB) {
            s_iq [tid] = q_data [(b0 + tid) * Ssz + t];
            s_iqa[tid] = qa_data[(b0 + tid) * Ssz + t];
        }
        __syncthreads();

        // ---- embedding gathers ----
        for (int p = tid; p < TB * Ksz; p += NT) {
            int b = p / Ksz, k = p % Ksz;
            s_qe[b][k] = q_embed_w[(size_t)s_iq[b] * Ksz + k];
        }
        for (int p = tid; p < TB * (Vsz / 4); p += NT) {   // qa as bf16 (GEMM A operand)
            int b = p / (Vsz / 4), v = (p % (Vsz / 4)) * 4;
            f4 val = *(const f4*)(qa_embed_w + (size_t)s_iqa[b] * Vsz + v);
            v4bf h; h[0] = (__bf16)val[0]; h[1] = (__bf16)val[1];
                    h[2] = (__bf16)val[2]; h[3] = (__bf16)val[3];
            *(v4bf*)&s_qab[(size_t)b * KPAD + v] = h;
        }
        __syncthreads();

        // ---- correlation logits: qe @ key_memory^T ----
        for (int p = tid; p < TB * Msz; p += NT) {
            int b = p / Msz, m = p % Msz;
            const float* km = s_km + m * Ksz;
            float acc = 0.f;
            #pragma unroll 5
            for (int k = 0; k < Ksz; ++k) acc += s_qe[b][k] * km[k];
            s_w[b][m] = acc;
        }
        __syncthreads();

        // ---- softmax (last wave, lanes 0-15) runs concurrent with WMMA waves ----
        if (wave == nwaves - 1 && lane < TB) {
            const int b = lane;
            float mx = -1e30f;
            for (int m = 0; m < Msz; ++m) mx = fmaxf(mx, s_w[b][m]);
            float sum = 0.f;
            for (int m = 0; m < Msz; ++m) { float e = __expf(s_w[b][m] - mx); s_w[b][m] = e; sum += e; }
            float inv = 1.f / sum;
            for (int m = 0; m < Msz; ++m) s_w[b][m] *= inv;
        }

        // ---- erase/add gates: D[16,200] = qa[16,200] @ W[200,200] via WMMA bf16 ----
        // 13 N-tiles x 2 GEMMs = 26 wave tasks; 7 k-steps of 32 (zero-padded); all LDS operands.
        for (int task = wave; task < 26; task += nwaves) {
            const int g   = task / 13;               // 0 = erase, 1 = add
            const int n0  = (task % 13) * 16;
            const int col = n0 + l16;
            const __bf16* wcol = s_Wt + (size_t)(g * NPAD + col) * KPAD; // W column, contiguous k
            const __bf16* arow = s_qab + (size_t)l16 * KPAD;             // qa row, contiguous k
            v8f acc = {};
            #pragma unroll
            for (int kk = 0; kk < KPAD / 32; ++kk) {
                const int kb = kk * 32;
                // A 16x32 bf16 layout: lane<16 K={kb..kb+7, kb+16..kb+23}; lane>=16 +8
                v8bf a0 = *(const v8bf*)(arow + kb + 8 * halfSel);
                v8bf a1 = *(const v8bf*)(arow + kb + 16 + 8 * halfSel);
                // B 32x16 bf16 layout: lane<16 K={kb..kb+7, kb+16..kb+23} @ N=l16; lane>=16 +8
                v8bf bb0 = *(const v8bf*)(wcol + kb + 8 * halfSel);
                v8bf bb1 = *(const v8bf*)(wcol + kb + 16 + 8 * halfSel);
                v16bf aF = __builtin_shufflevector(a0, a1, 0,1,2,3,4,5,6,7,8,9,10,11,12,13,14,15);
                v16bf bF = __builtin_shufflevector(bb0, bb1, 0,1,2,3,4,5,6,7,8,9,10,11,12,13,14,15);
                acc = __builtin_amdgcn_wmma_f32_16x16x32_bf16(
                          false, aF, false, bF, (short)0, acc, false, false);
            }
            if (col < Vsz) {
                float bv = (g ? add_b : erase_b)[col];
                float (*dst)[Vsz] = g ? s_a : s_e;
                #pragma unroll
                for (int r = 0; r < 8; ++r) {        // C/D: VGPR r -> M=r (lo half) / 8+r (hi half)
                    float vv = acc[r] + bv;
                    dst[r + halfSel * 8][col] = g ? tanhf(vv) : sigmoidf_(vv);
                }
            }
        }
        __syncthreads();

        // ---- fused pass over Mv (float4): read accumulation + erase/add update ----
        for (int p = tid; p < TB * (Vsz / 4); p += NT) {
            int b = p / (Vsz / 4), v = (p % (Vsz / 4)) * 4;
            const float* wb = s_w[b];
            f4 e4 = *(const f4*)&s_e[b][v];
            f4 a4 = *(const f4*)&s_a[b][v];
            f4* mv = (f4*)(Mv + (size_t)(b0 + b) * (Msz * Vsz) + v);
            f4 acc = {};
            #pragma unroll 5
            for (int m = 0; m < Msz; ++m) {
                f4 x = mv[m * (Vsz / 4)];
                float wm = wb[m];
                acc += wm * x;
                mv[m * (Vsz / 4)] = x * (1.f - wm * e4) + wm * a4;
            }
            *(f4*)&s_rd[b][v] = acc;
        }
        __syncthreads();

        // ---- summary head: tanh([read, qe] @ summary_W + b) ----
        for (int p = tid; p < TB * Fsz; p += NT) {
            int b = p / Fsz, f = p % Fsz;
            float acc = summary_b[f];
            for (int j = 0; j < Vsz; ++j) acc += s_rd[b][j] * s_sumW[j * Fsz + f];
            for (int k = 0; k < Ksz; ++k) acc += s_qe[b][k] * s_sumW[(Vsz + k) * Fsz + f];
            s_s[b][f] = tanhf(acc);
        }
        __syncthreads();

        // ---- IRT scalars + outputs ----
        if (tid < TB) {
            const int b = tid;
            float th = theta_b[0], be = beta_b[0], al = alpha_b[0];
            for (int f = 0; f < Fsz; ++f) { th += s_s[b][f] * theta_W[f]; al += s_s[b][f] * alpha_W[f]; }
            for (int k = 0; k < Ksz; ++k) { be += s_qe[b][k] * beta_W[k]; al += s_qe[b][k] * alpha_W[Fsz + k]; }
            be = tanhf(be);
            al = softplusf_(al);
            float z  = al * (3.0f * th - be);
            float pp = sigmoidf_(z);
            const size_t BS = (size_t)Bsz * Ssz;
            const size_t o  = (size_t)(b0 + b) * Ssz + t;
            out[o]          = pp;
            out[BS + o]     = th;
            out[2 * BS + o] = be;
            out[3 * BS + o] = al;
            out[4 * BS + o] = z;
        }
        __syncthreads();
    }
}

extern "C" void kernel_launch(void* const* d_in, const int* in_sizes, int n_in,
                              void* d_out, int out_size, void* d_ws, size_t ws_size,
                              hipStream_t stream) {
    deep_irt_kernel<<<Bsz / TB, NT, DYN_SMEM, stream>>>(
        (const int*)d_in[0],  (const int*)d_in[1],
        (const float*)d_in[2],  (const float*)d_in[3],
        (const float*)d_in[4],  (const float*)d_in[5],
        (const float*)d_in[6],  (const float*)d_in[7],
        (const float*)d_in[8],  (const float*)d_in[9],
        (const float*)d_in[10], (const float*)d_in[11],
        (const float*)d_in[12], (const float*)d_in[13],
        (const float*)d_in[14], (const float*)d_in[15],
        (const float*)d_in[16], (const float*)d_in[17],
        (float*)d_ws,           // needs B*M*V*4 = 81.92 MB of scratch
        (float*)d_out);
}
